// EGNNLayerLN_33663953666609
// MI455X (gfx1250) — matrix operand order repaired and checked by reading, not compiled, not run on hardware
//
#include <hip/hip_runtime.h>
#include <hip/hip_fp16.h>
#include <hip/hip_bf16.h>

typedef __attribute__((ext_vector_type(16))) _Float16 v16h;
typedef __attribute__((ext_vector_type(8)))  float    v8f;

#define B_  4
#define N_  256
#define H_  128
#define DE_ 9
#define JSPLIT 8
#define JSEG (N_/JSPLIT)   // 32 j's per block

union FragH {
  v16h h;
  uint4 q[2];
};
union Pack8 {
  uint4 q;
  _Float16 x[8];
};

__device__ __forceinline__ float silu_f(float v) {
  return v / (1.0f + __expf(-v));
}

// ---- CDNA5 async global->LDS copy (ASYNCcnt-tracked, no VGPR round trip) ----
// LDS address operand = low 32 bits of the generic pointer (ISA: flat->LDS
// aperture mapping discards upper bits; async op adds LDS_BASE itself).
__device__ __forceinline__ void async_ld_b128(void* lds_dst, const void* gsrc) {
  const unsigned int ldsa = (unsigned int)(unsigned long long)lds_dst;
  const unsigned long long ga = (unsigned long long)gsrc;
  asm volatile("global_load_async_to_lds_b128 %0, %1, off"
               :: "v"(ldsa), "v"(ga) : "memory");
}
__device__ __forceinline__ void wait_async0() {
  asm volatile("s_wait_asynccnt 0" ::: "memory");
}

// ---------------- LDS layout (dynamic shared) ----------------
constexpr int SM_P    = 0;                  // 256x128 f16  (pre1 / hc, reused)
constexpr int SM_MH   = SM_P    + 65536;    // 256x128 f16  (m)
constexpr int SM_W2   = SM_MH   + 65536;    // 128x128 f16  (w2m^T [n][k])
constexpr int SM_W1C  = SM_W2   + 32768;    // 128x128 f16  (w1c^T [n][k])
constexpr int SM_AI   = SM_W1C  + 32768;    // 16x128 f32
constexpr int SM_AJ0  = SM_AI   + 8192;     // 16x128 f32 (ping)
constexpr int SM_AJ1  = SM_AJ0  + 8192;     // 16x128 f32 (pong)
constexpr int SM_WT   = SM_AJ1  + 8192;     // 9x128 f32 (w1m tail)
constexpr int SM_B2M  = SM_WT   + 4608;     // 128 f32
constexpr int SM_B1C  = SM_B2M  + 512;      // 128 f32
constexpr int SM_W2C  = SM_B1C  + 512;      // 128 f32
constexpr int SM_MS   = SM_W2C  + 512;      // 16x128 f32 (msum accumulator)
constexpr int SM_DL   = SM_MS   + 8192;     // 48 f32 (delta accumulator)
constexpr int SM_RIJ  = SM_DL   + 256;      // 256x3 f32
constexpr int SM_CI   = SM_RIJ  + 3072;     // 48 f32
constexpr int SM_CJ   = SM_CI   + 256;      // 48 f32
constexpr int SMEM_BYTES = SM_CJ + 256;     // 239360 bytes (< 320KB/WGP)

// ---------------- prologue kernels ----------------

// Ai[bn][h] = x[bn] . w1m[0:128, h] ; Aj[bn][h] = x[bn] . w1m[128:256, h] + b1m[h]
__global__ void __launch_bounds__(128)
egnn_prep_nodes(const float* __restrict__ x, const float* __restrict__ w1m,
                const float* __restrict__ b1m,
                float* __restrict__ Ai, float* __restrict__ Aj) {
  const int bn = blockIdx.x;
  const int h  = threadIdx.x;
  __shared__ float xr[H_];
  xr[h] = x[(size_t)bn * H_ + h];
  __syncthreads();
  float a1 = 0.f, a2 = 0.f;
  #pragma unroll 4
  for (int k = 0; k < H_; ++k) {
    const float xv = xr[k];
    a1 += xv * w1m[k * H_ + h];
    a2 += xv * w1m[(H_ + k) * H_ + h];
  }
  Ai[(size_t)bn * H_ + h] = a1;
  Aj[(size_t)bn * H_ + h] = a2 + b1m[h];
}

// transpose + f16-convert w2m, w1c:  wT[n][k] = (f16) w[k][n]
__global__ void __launch_bounds__(256)
egnn_prep_weights(const float* __restrict__ w2m, const float* __restrict__ w1c,
                  _Float16* __restrict__ w2mT, _Float16* __restrict__ w1cT) {
  const int idx = blockIdx.x * blockDim.x + threadIdx.x;
  if (idx < H_ * H_) {
    const int n = idx >> 7, k = idx & (H_ - 1);
    w2mT[idx] = (_Float16)w2m[k * H_ + n];
    w1cT[idx] = (_Float16)w1c[k * H_ + n];
  }
}

__global__ void __launch_bounds__(256)
egnn_zero(float* __restrict__ msum, float* __restrict__ delta) {
  const int idx = blockIdx.x * blockDim.x + threadIdx.x;
  if (idx < B_ * N_ * H_) msum[idx] = 0.f;
  if (idx < B_ * N_ * 3)  delta[idx] = 0.f;
}

// ---------------- fused edge kernel (WMMA + async LDS staging) ----------------
__global__ void __launch_bounds__(256)
egnn_edge(const float* __restrict__ coords,
          const float* __restrict__ Ai, const float* __restrict__ Aj,
          const float* __restrict__ w1m,
          const float* __restrict__ b2m, const float* __restrict__ b1c,
          const float* __restrict__ w2c, const float* __restrict__ b2c,
          const _Float16* __restrict__ w2mT, const _Float16* __restrict__ w1cT,
          float* __restrict__ msum_g, float* __restrict__ delta_g) {
  extern __shared__ __align__(16) char smem[];
  _Float16* P    = (_Float16*)(smem + SM_P);
  _Float16* Mh   = (_Float16*)(smem + SM_MH);
  _Float16* sW2  = (_Float16*)(smem + SM_W2);
  _Float16* sW1c = (_Float16*)(smem + SM_W1C);
  float* sAi   = (float*)(smem + SM_AI);
  float* sAj0  = (float*)(smem + SM_AJ0);
  float* sAj1  = (float*)(smem + SM_AJ1);
  float* sWt   = (float*)(smem + SM_WT);
  float* sB2m  = (float*)(smem + SM_B2M);
  float* sB1c  = (float*)(smem + SM_B1C);
  float* sW2c  = (float*)(smem + SM_W2C);
  float* sMsum = (float*)(smem + SM_MS);
  float* sDelta= (float*)(smem + SM_DL);
  float* sRij  = (float*)(smem + SM_RIJ);
  float* sCi   = (float*)(smem + SM_CI);
  float* sCj   = (float*)(smem + SM_CJ);

  const int tid = threadIdx.x;
  const int bx  = blockIdx.x;
  const int b   = bx / (16 * JSPLIT);
  const int rem = bx % (16 * JSPLIT);
  const int it  = rem / JSPLIT;
  const int js  = rem % JSPLIT;
  const int i0  = it * 16;
  const int j0  = js * JSEG;
  const float b2c0 = b2c[0];

  // ---- stage weights + Ai + first Aj tile via async global->LDS
  {
    const uint4* s2 = (const uint4*)w2mT;
    const uint4* s1 = (const uint4*)w1cT;
    uint4* d2 = (uint4*)sW2;
    uint4* d1 = (uint4*)sW1c;
    for (int idx = tid; idx < (H_ * H_) / 8; idx += 256) {
      async_ld_b128(d2 + idx, s2 + idx);
      async_ld_b128(d1 + idx, s1 + idx);
    }
    const uint4* sa = (const uint4*)(Ai + (size_t)(b * N_ + i0) * H_);
    uint4* da = (uint4*)sAi;
    for (int idx = tid; idx < (16 * H_) / 4; idx += 256)
      async_ld_b128(da + idx, sa + idx);
    const uint4* sj = (const uint4*)(Aj + (size_t)(b * N_ + j0) * H_);
    uint4* dj = (uint4*)sAj0;
    for (int idx = tid; idx < (16 * H_) / 4; idx += 256)
      async_ld_b128(dj + idx, sj + idx);
  }
  // synchronous tiny stages
  for (int idx = tid; idx < DE_ * H_; idx += 256) sWt[idx] = w1m[2 * H_ * H_ + idx];
  if (tid < H_) { sB2m[tid] = b2m[tid]; sB1c[tid] = b1c[tid]; sW2c[tid] = w2c[tid]; }
  for (int idx = tid; idx < 16 * H_; idx += 256) sMsum[idx] = 0.f;
  if (tid < 48) { sCi[tid] = coords[(size_t)(b * N_ + i0) * 3 + tid]; sDelta[tid] = 0.f; }

  const int lane  = tid & 31;
  const int wave  = tid >> 5;
  const int lhalf = lane >> 4;   // 0: lanes 0-15, 1: lanes 16-31
  const int lmod  = lane & 15;

  for (int jt = 0; jt < JSEG; jt += 16) {
    const int jb  = j0 + jt;
    const int cur = (jt >> 4) & 1;
    const float* sAj = cur ? sAj1 : sAj0;

    if (tid < 48) sCj[tid] = coords[(size_t)(b * N_ + jb) * 3 + tid];
    wait_async0();      // my async stores to LDS (incl. current Aj buffer) done
    __syncthreads();    // everyone done; staged data + sCj visible

    // kick off the NEXT Aj tile asynchronously; hidden behind ~1K WMMAs below
    if (jt + 16 < JSEG) {
      const uint4* sj = (const uint4*)(Aj + (size_t)(b * N_ + jb + 16) * H_);
      uint4* dj = (uint4*)(cur ? sAj0 : sAj1);
      for (int idx = tid; idx < (16 * H_) / 4; idx += 256)
        async_ld_b128(dj + idx, sj + idx);
    }

    // ---- phase 1: distance embedding + precomputed layer-1 parts + silu -> P (f16)
    {
      const int e  = tid;          // edge = il*16 + jl
      const int il = e >> 4;
      const int jl = e & 15;
      const float rx = sCi[il * 3 + 0] - sCj[jl * 3 + 0];
      const float ry = sCi[il * 3 + 1] - sCj[jl * 3 + 1];
      const float rz = sCi[il * 3 + 2] - sCj[jl * 3 + 2];
      const float sq = rx * rx + ry * ry + rz * rz;
      const float d  = sq > 0.f ? sqrtf(sq) : 0.f;
      sRij[e * 3 + 0] = rx; sRij[e * 3 + 1] = ry; sRij[e * 3 + 2] = rz;
      float emb[DE_];
      emb[0] = d;
      #pragma unroll
      for (int k = 0; k < 4; ++k) {
        const float ang = fminf(d * (float)(k + 1), 100.f);  // d >= 0: upper clamp only
        emb[1 + k] = __sinf(ang);
        emb[5 + k] = __cosf(ang);
      }
      const float* ai = sAi + il * H_;
      const float* aj = sAj + jl * H_;
      const int hstart = (e & 7) << 4;   // stagger LDS banks
      for (int hh = 0; hh < H_; ++hh) {
        const int h = (hh + hstart) & (H_ - 1);
        float v = ai[h] + aj[h];
        #pragma unroll
        for (int k = 0; k < DE_; ++k) v += emb[k] * sWt[k * H_ + h];
        P[e * H_ + h] = (_Float16)silu_f(v);
      }
    }
    __syncthreads();

    // ---- phase 2 (WMMA): m = P @ w2m^T + b2m ; accumulate msum ; write Mh (f16)
    {
      const int nt = wave;  // each wave owns one 16-wide output column tile
      FragH fb[4];
      #pragma unroll
      for (int ks = 0; ks < 4; ++ks) {
        const _Float16* rb = sW2 + (nt * 16 + lmod) * H_ + ks * 32 + (lhalf ? 8 : 0);
        fb[ks].q[0] = *(const uint4*)(rb);
        fb[ks].q[1] = *(const uint4*)(rb + 16);
      }
      for (int mt = 0; mt < 16; ++mt) {
        FragH fa[4];
        #pragma unroll
        for (int ks = 0; ks < 4; ++ks) {   // issue all 8 b128 loads up front
          const _Float16* ra = P + (mt * 16 + lmod) * H_ + ks * 32 + (lhalf ? 8 : 0);
          fa[ks].q[0] = *(const uint4*)(ra);
          fa[ks].q[1] = *(const uint4*)(ra + 16);
        }
        v8f acc = {0.f, 0.f, 0.f, 0.f, 0.f, 0.f, 0.f, 0.f};
        #pragma unroll
        for (int ks = 0; ks < 4; ++ks)
          acc = __builtin_amdgcn_wmma_f32_16x16x32_f16(false, fa[ks].h, false, fb[ks].h,
                                                       (short)0, acc, false, false);
        const int hcol = nt * 16 + lmod;
        const float bias = sB2m[hcol];
        float sum = 0.f;
        #pragma unroll
        for (int rr = 0; rr < 8; ++rr) {
          const float v = acc[rr] + bias;
          sum += v;
          const int jl = rr + (lhalf << 3);
          Mh[(mt * 16 + jl) * H_ + hcol] = (_Float16)v;
        }
        atomicAdd(&sMsum[mt * H_ + hcol], sum);   // ds_add_f32
      }
    }
    __syncthreads();

    // ---- phase 3 (WMMA): hc = silu(m @ w1c^T + b1c) -> reuse P
    {
      const int nt = wave;
      FragH fb[4];
      #pragma unroll
      for (int ks = 0; ks < 4; ++ks) {
        const _Float16* rb = sW1c + (nt * 16 + lmod) * H_ + ks * 32 + (lhalf ? 8 : 0);
        fb[ks].q[0] = *(const uint4*)(rb);
        fb[ks].q[1] = *(const uint4*)(rb + 16);
      }
      for (int mt = 0; mt < 16; ++mt) {
        FragH fa[4];
        #pragma unroll
        for (int ks = 0; ks < 4; ++ks) {
          const _Float16* ra = Mh + (mt * 16 + lmod) * H_ + ks * 32 + (lhalf ? 8 : 0);
          fa[ks].q[0] = *(const uint4*)(ra);
          fa[ks].q[1] = *(const uint4*)(ra + 16);
        }
        v8f acc = {0.f, 0.f, 0.f, 0.f, 0.f, 0.f, 0.f, 0.f};
        #pragma unroll
        for (int ks = 0; ks < 4; ++ks)
          acc = __builtin_amdgcn_wmma_f32_16x16x32_f16(false, fa[ks].h, false, fb[ks].h,
                                                       (short)0, acc, false, false);
        const int hcol = nt * 16 + lmod;
        const float bias = sB1c[hcol];
        #pragma unroll
        for (int rr = 0; rr < 8; ++rr) {
          const float v = acc[rr] + bias;
          const int jl = rr + (lhalf << 3);
          P[(mt * 16 + jl) * H_ + hcol] = (_Float16)silu_f(v);
        }
      }
    }
    __syncthreads();

    // ---- phase 4: coord_coef = tanh(hc . w2c + b2c), accumulate delta
    {
      const int e = tid;
      const _Float16* row = P + e * H_;
      float dotv = 0.f;
      for (int h8 = 0; h8 < H_; h8 += 8) {
        Pack8 u;
        u.q = *(const uint4*)(row + h8);
        #pragma unroll
        for (int k = 0; k < 8; ++k) dotv += (float)u.x[k] * sW2c[h8 + k];
      }
      const float cc = tanhf(dotv + b2c0);
      const int il = e >> 4;
      atomicAdd(&sDelta[il * 3 + 0], cc * sRij[e * 3 + 0]);
      atomicAdd(&sDelta[il * 3 + 1], cc * sRij[e * 3 + 1]);
      atomicAdd(&sDelta[il * 3 + 2], cc * sRij[e * 3 + 2]);
    }
    // no trailing barrier: next loop-top barrier (after wait_async0) covers WAR
  }
  __syncthreads();

  // flush block partials
  for (int idx = tid; idx < 16 * H_; idx += 256)
    atomicAdd(&msum_g[(size_t)(b * N_ + i0) * H_ + idx], sMsum[idx]);
  if (tid < 48)
    atomicAdd(&delta_g[(size_t)(b * N_ + i0) * 3 + tid], sDelta[tid]);
}

// ---------------- epilogue: feature MLP + LayerNorm + coords ----------------
__global__ void __launch_bounds__(128)
egnn_final(const float* __restrict__ x, const float* __restrict__ coords,
           const float* __restrict__ msum, const float* __restrict__ delta,
           const float* __restrict__ w1f, const float* __restrict__ b1f,
           const float* __restrict__ w2f, const float* __restrict__ b2f,
           const float* __restrict__ ln_g, const float* __restrict__ ln_b,
           float* __restrict__ out) {
  const int bn = blockIdx.x;
  const int h  = threadIdx.x;
  __shared__ float sm[H_];
  __shared__ float st[H_];
  __shared__ float red[H_];

  sm[h] = msum[(size_t)bn * H_ + h];
  __syncthreads();

  float a = b1f[h];
  #pragma unroll 4
  for (int k = 0; k < H_; ++k) a += sm[k] * w1f[k * H_ + h];
  st[h] = silu_f(a);
  __syncthreads();

  float v = b2f[h];
  #pragma unroll 4
  for (int k = 0; k < H_; ++k) v += st[k] * w2f[k * H_ + h];
  v += x[(size_t)bn * H_ + h];

  red[h] = v;
  __syncthreads();
  for (int off = 64; off > 0; off >>= 1) {
    if (h < off) red[h] += red[h + off];
    __syncthreads();
  }
  const float mu = red[0] * (1.0f / H_);
  __syncthreads();
  const float dv = v - mu;
  red[h] = dv * dv;
  __syncthreads();
  for (int off = 64; off > 0; off >>= 1) {
    if (h < off) red[h] += red[h + off];
    __syncthreads();
  }
  const float var = red[0] * (1.0f / H_);
  out[(size_t)bn * H_ + h] = dv * rsqrtf(var + 1e-5f) * ln_g[h] + ln_b[h];

  if (h < 3) {
    out[(size_t)B_ * N_ * H_ + bn * 3 + h] =
        coords[(size_t)bn * 3 + h] + 0.01f * delta[(size_t)bn * 3 + h];
  }
}

// ---------------- launcher ----------------
extern "C" void kernel_launch(void* const* d_in, const int* in_sizes, int n_in,
                              void* d_out, int out_size, void* d_ws, size_t ws_size,
                              hipStream_t stream) {
  const float* x      = (const float*)d_in[0];
  const float* coords = (const float*)d_in[1];
  const float* w1m    = (const float*)d_in[2];
  const float* b1m    = (const float*)d_in[3];
  const float* w2m    = (const float*)d_in[4];
  const float* b2m    = (const float*)d_in[5];
  const float* w1c    = (const float*)d_in[6];
  const float* b1c    = (const float*)d_in[7];
  const float* w2c    = (const float*)d_in[8];
  const float* b2c    = (const float*)d_in[9];
  const float* w1f    = (const float*)d_in[10];
  const float* b1f    = (const float*)d_in[11];
  const float* w2f    = (const float*)d_in[12];
  const float* b2f    = (const float*)d_in[13];
  const float* ln_g   = (const float*)d_in[14];
  const float* ln_b   = (const float*)d_in[15];
  float* out = (float*)d_out;

  char* ws = (char*)d_ws;
  float*    Ai    = (float*)(ws);                  // 524288 B
  float*    Aj    = (float*)(ws + 524288);         // 524288 B (includes +b1m)
  float*    msum  = (float*)(ws + 1048576);        // 524288 B
  float*    delta = (float*)(ws + 1572864);        // 12288 B
  _Float16* w2mT  = (_Float16*)(ws + 1585152);     // 32768 B
  _Float16* w1cT  = (_Float16*)(ws + 1617920);     // 32768 B

  egnn_prep_nodes<<<B_ * N_, H_, 0, stream>>>(x, w1m, b1m, Ai, Aj);
  egnn_prep_weights<<<(H_ * H_ + 255) / 256, 256, 0, stream>>>(w2m, w1c, w2mT, w1cT);
  egnn_zero<<<(B_ * N_ * H_ + 255) / 256, 256, 0, stream>>>(msum, delta);
  egnn_edge<<<B_ * 16 * JSPLIT, 256, SMEM_BYTES, stream>>>(
      coords, Ai, Aj, w1m, b2m, b1c, w2c, b2c, w2mT, w1cT, msum, delta);
  egnn_final<<<B_ * N_, H_, 0, stream>>>(x, coords, msum, delta, w1f, b1f, w2f, b2f,
                                         ln_g, ln_b, out);
}
